// NonLocalBlock2D_42253888258169
// MI455X (gfx1250) — matrix operand order repaired and checked by reading, not compiled, not run on hardware
//
#include <hip/hip_runtime.h>
#include <hip/hip_bf16.h>

// ---------------------------------------------------------------------------
// NonLocalBlock2D for MI455X (gfx1250): bf16 WMMA projections + flash attention
// x:[8,256,64,64] f32 -> out:[8,128,64,64] f32
// ---------------------------------------------------------------------------

#define B_SZ   8
#define C_IN   256
#define C_INT  128
#define NPOS   4096   // 64*64

typedef __attribute__((ext_vector_type(16))) __bf16 v16bf;
typedef __attribute__((ext_vector_type(2)))  __bf16 v2bf;
typedef __attribute__((ext_vector_type(8)))  float  v8f;
typedef __attribute__((ext_vector_type(2)))  float  v2f;
typedef int v4i __attribute__((vector_size(16)));

#define AS1 __attribute__((address_space(1)))
#define AS3 __attribute__((address_space(3)))

union AV { v16bf v; unsigned int u[8]; };

static __device__ __forceinline__ unsigned int pack2(unsigned short lo, unsigned short hi) {
    return (unsigned int)lo | ((unsigned int)hi << 16);
}

// Packed f32x2 -> bf16x2: fptrunc pair; backend selects v_cvt_pk_bf16_f32 when
// the target has it, otherwise emits the RNE software sequence.
static __device__ __forceinline__ unsigned int pk_bf16(float a, float b) {
    v2f t = {a, b};
    v2bf r = __builtin_convertvector(t, v2bf);
    return __builtin_bit_cast(unsigned int, r);
}
static __device__ __forceinline__ unsigned int pk_bf16v(v2f t) {
    v2bf r = __builtin_convertvector(t, v2bf);
    return __builtin_bit_cast(unsigned int, r);
}
static __device__ __forceinline__ unsigned short f2bf(float f) {
    __bf16 h = (__bf16)f;                       // hw cvt if available (RNE)
    return __builtin_bit_cast(unsigned short, h);
}

static __device__ __forceinline__ v8f wmma_bf16(const v16bf& a, const v16bf& b, const v8f& c) {
    return __builtin_amdgcn_wmma_f32_16x16x32_bf16(false, a, false, b, (short)0, c, false, false);
}

#if __has_builtin(__builtin_amdgcn_global_load_async_to_lds_b128) && \
    __has_builtin(__builtin_amdgcn_s_wait_asynccnt)
#define USE_ASYNC_LDS 1
#else
#define USE_ASYNC_LDS 0
#endif

// ---------------------------------------------------------------------------
// Kernel 1: projections theta/phi/g = x^T W^T + b, output bf16 [3][B][N][128]
// grid: 3 * B * (N/128) blocks of 256 threads (8 waves); wave = 16 positions.
// ---------------------------------------------------------------------------
__global__ __launch_bounds__(256) void nlb_proj_kernel(
    const float* __restrict__ x,
    const float* __restrict__ w_theta, const float* __restrict__ b_theta,
    const float* __restrict__ w_phi,   const float* __restrict__ b_phi,
    const float* __restrict__ w_g,     const float* __restrict__ b_g,
    unsigned short* __restrict__ ws_proj)
{
    __shared__ unsigned short wsh[C_INT * C_IN];   // 64 KB bf16 weights

    int blk   = blockIdx.x;
    int proj  = blk % 3;
    int rest  = blk / 3;
    int b     = rest % B_SZ;
    int ptile = rest / B_SZ;                       // 0..31: 128-position tile

    const float* W    = (proj == 0) ? w_theta : (proj == 1) ? w_phi : w_g;
    const float* bias = (proj == 0) ? b_theta : (proj == 1) ? b_phi : b_g;

    {   // stage weights as bf16, packed pair conversion (vector loads)
        unsigned int* wsh32 = (unsigned int*)wsh;
        const v2f* W2 = (const v2f*)W;
        for (int i = threadIdx.x; i < (C_INT * C_IN) / 2; i += 256)
            wsh32[i] = pk_bf16v(W2[i]);
    }
    __syncthreads();

    int lane = threadIdx.x & 31;
    int wv   = threadIdx.x >> 5;
    int m    = lane & 15;          // row within 16 (A) / column within 16 (B,D)
    int half = lane >> 4;
    int kbA  = half * 16;          // A-layout K base (contiguous, matches B)
    int kbB  = half * 16;          // B-layout K base (contiguous, matches A)
    int p0   = ptile * 128 + wv * 16;

    const float* xb = x + (size_t)b * C_IN * NPOS;

    v8f acc[8];
    #pragma unroll
    for (int t = 0; t < 8; ++t) acc[t] = (v8f){};

    #pragma unroll 2
    for (int kc = 0; kc < 8; ++kc) {               // K = 256 in chunks of 32
        int c0 = kc * 32;
        // A tile: 16 positions x 32 input channels (channel-major x => strided)
        AV A;
        const float* xp = xb + (p0 + m);
        #pragma unroll
        for (int j = 0; j < 4; ++j) {
            int c  = c0 + kbA + 2 * j;
            int c2 = c0 + 8 + kbA + 2 * j;
            A.u[j]     = pk_bf16(xp[(size_t)c  * NPOS], xp[(size_t)(c  + 1) * NPOS]);
            A.u[4 + j] = pk_bf16(xp[(size_t)c2 * NPOS], xp[(size_t)(c2 + 1) * NPOS]);
        }
        #pragma unroll
        for (int dt = 0; dt < 8; ++dt) {           // 8 output-channel tiles
            AV Bm;
            int d = dt * 16 + m;
            const unsigned short* wrow = &wsh[d * C_IN + c0 + kbB];
            #pragma unroll
            for (int j = 0; j < 8; ++j)
                Bm.u[j] = *(const unsigned int*)(wrow + 2 * j);
            acc[dt] = wmma_bf16(A.v, Bm.v, acc[dt]);
        }
    }

    // bias + store bf16 into workspace [proj][b][pos][d]
    unsigned short* out = ws_proj + ((size_t)proj * B_SZ + b) * NPOS * C_INT;
    int mo = 8 * half;
    #pragma unroll
    for (int dt = 0; dt < 8; ++dt) {
        int d = dt * 16 + m;
        float bv = bias[d];
        #pragma unroll
        for (int r = 0; r < 8; ++r) {
            int p = p0 + r + mo;
            out[(size_t)p * C_INT + d] = f2bf(acc[dt][r] + bv);
        }
    }
}

// ---------------------------------------------------------------------------
// Kernel 2: flash attention  Y = softmax(theta phi^T) g, out f32 [B][128][N]
// grid: B * (N/128) blocks of 256 threads; each wave owns a 16-query tile.
// ---------------------------------------------------------------------------
__global__ __launch_bounds__(256) void nlb_attn_kernel(
    const unsigned short* __restrict__ theta,
    const unsigned short* __restrict__ phi,
    const unsigned short* __restrict__ g,
    float* __restrict__ out)
{
    __shared__ unsigned short phi_s[32 * C_INT];   // 8 KB key chunk
    __shared__ unsigned short g_s[32 * C_INT];     // 8 KB value chunk
    __shared__ float p_s[8][16 * 32];              // 16 KB per-wave P transpose

    int b    = blockIdx.x >> 5;                    // 8 batches
    int qblk = blockIdx.x & 31;                    // 32 query blocks of 128
    int lane = threadIdx.x & 31;
    int wv   = threadIdx.x >> 5;
    int m    = lane & 15;
    int half = lane >> 4;
    int kbA  = half * 16;
    int kbB  = half * 16;
    int q0   = qblk * 128 + wv * 16;

    const unsigned short* thb = theta + (size_t)b * NPOS * C_INT;
    const unsigned short* phb = phi   + (size_t)b * NPOS * C_INT;
    const unsigned short* gb  = g     + (size_t)b * NPOS * C_INT;

    // theta tile (16 x 128) held in registers, A layout, 4 K-chunks of 32
    AV Ath[4];
    #pragma unroll
    for (int kc = 0; kc < 4; ++kc) {
        const unsigned short* trow = thb + (size_t)(q0 + m) * C_INT + kc * 32 + kbA;
        #pragma unroll
        for (int j = 0; j < 4; ++j) {
            Ath[kc].u[j]     = *(const unsigned int*)(trow + 2 * j);
            Ath[kc].u[4 + j] = *(const unsigned int*)(trow + 8 + 2 * j);
        }
    }

    v8f O[8];
    #pragma unroll
    for (int t = 0; t < 8; ++t) O[t] = (v8f){};
    float mrow[8], lrow[8];
    #pragma unroll
    for (int r = 0; r < 8; ++r) { mrow[r] = -3.0e38f; lrow[r] = 0.0f; }

    float* ps = p_s[wv];

    for (int k0 = 0; k0 < NPOS; k0 += 32) {
        __syncthreads();                           // protect LDS from prev iter

        // ---- stage phi/g chunk (32 keys x 128 ch bf16 = 8 KB each) --------
#if USE_ASYNC_LDS
        {
            char* gp = (char*)(phb + (size_t)k0 * C_INT);
            char* gg = (char*)(gb  + (size_t)k0 * C_INT);
            #pragma unroll
            for (int pass = 0; pass < 2; ++pass) {
                int off = ((int)threadIdx.x + pass * 256) * 16;   // 4 KB / pass
                __builtin_amdgcn_global_load_async_to_lds_b128(
                    (AS1 v4i*)(gp + off),
                    (AS3 v4i*)((char*)phi_s + off), 0, 0);
                __builtin_amdgcn_global_load_async_to_lds_b128(
                    (AS1 v4i*)(gg + off),
                    (AS3 v4i*)((char*)g_s + off), 0, 0);
            }
            __builtin_amdgcn_s_wait_asynccnt(0);
        }
#else
        {
            const unsigned int* psrc = (const unsigned int*)(phb + (size_t)k0 * C_INT);
            const unsigned int* gsrc = (const unsigned int*)(gb  + (size_t)k0 * C_INT);
            unsigned int* pd = (unsigned int*)phi_s;
            unsigned int* gd = (unsigned int*)g_s;
            for (int i = threadIdx.x; i < 2048; i += 256) { pd[i] = psrc[i]; gd[i] = gsrc[i]; }
        }
#endif
        if (k0 + 32 < NPOS) {                      // gfx1250 global_prefetch hints
            __builtin_prefetch(phb + (size_t)(k0 + 32) * C_INT + threadIdx.x * 16, 0, 1);
            __builtin_prefetch(gb  + (size_t)(k0 + 32) * C_INT + threadIdx.x * 16, 0, 1);
        }
        __syncthreads();

        // ---- S = theta * phi^T : two 16-key tiles, K = 128 ----------------
        v8f S0 = (v8f){}, S1 = (v8f){};
        #pragma unroll
        for (int kc = 0; kc < 4; ++kc) {
            AV Bp;
            const unsigned short* prow = phi_s + (0 * 16 + m) * C_INT + kc * 32 + kbB;
            #pragma unroll
            for (int j = 0; j < 8; ++j) Bp.u[j] = *(const unsigned int*)(prow + 2 * j);
            S0 = wmma_bf16(Ath[kc].v, Bp.v, S0);
        }
        #pragma unroll
        for (int kc = 0; kc < 4; ++kc) {
            AV Bp;
            const unsigned short* prow = phi_s + (1 * 16 + m) * C_INT + kc * 32 + kbB;
            #pragma unroll
            for (int j = 0; j < 8; ++j) Bp.u[j] = *(const unsigned int*)(prow + 2 * j);
            S1 = wmma_bf16(Ath[kc].v, Bp.v, S1);
        }

        // ---- online softmax; rows live across 16-lane halves (wave32) -----
        #pragma unroll
        for (int r = 0; r < 8; ++r) {
            float v = fmaxf(S0[r], S1[r]);
            v = fmaxf(v, __shfl_xor(v, 1, 32));
            v = fmaxf(v, __shfl_xor(v, 2, 32));
            v = fmaxf(v, __shfl_xor(v, 4, 32));
            v = fmaxf(v, __shfl_xor(v, 8, 32));
            float nm = fmaxf(mrow[r], v);

            float pa = __expf(S0[r] - nm);
            float pb = __expf(S1[r] - nm);
            int q = r + 8 * half;
            ps[q * 32 + m]      = pa;              // scatter P for transpose
            ps[q * 32 + 16 + m] = pb;

            float s = pa + pb;
            s += __shfl_xor(s, 1, 32);
            s += __shfl_xor(s, 2, 32);
            s += __shfl_xor(s, 4, 32);
            s += __shfl_xor(s, 8, 32);

            float alpha = __expf(mrow[r] - nm);
            lrow[r] = lrow[r] * alpha + s;
            mrow[r] = nm;
            #pragma unroll
            for (int t = 0; t < 8; ++t) O[t][r] *= alpha;
        }

        // ---- read P back in A layout (row = query, K = 32 keys), bf16 -----
        AV Ap;
        #pragma unroll
        for (int j = 0; j < 4; ++j) {
            int kk = kbA + 2 * j;
            v2f t0 = *(const v2f*)&ps[m * 32 + kk];        // 8B-aligned (kk even)
            v2f t1 = *(const v2f*)&ps[m * 32 + 8 + kk];
            Ap.u[j]     = pk_bf16v(t0);
            Ap.u[4 + j] = pk_bf16v(t1);
        }

        // ---- O += P * G : 8 output-channel tiles, K = 32 keys -------------
        #pragma unroll
        for (int dt = 0; dt < 8; ++dt) {
            AV Bg;
            int d = dt * 16 + m;
            #pragma unroll
            for (int j = 0; j < 8; ++j) {
                int kk = kbB + 2 * j;
                Bg.u[j] = pack2(g_s[kk * C_INT + d], g_s[(kk + 1) * C_INT + d]);
            }
            O[dt] = wmma_bf16(Ap.v, Bg.v, O[dt]);
        }
    }

    // finalize: divide by row sums, store out[b][d][q] (f32, channel-major)
    float* ob = out + (size_t)b * C_INT * NPOS;
    float rinv[8];
    #pragma unroll
    for (int r = 0; r < 8; ++r) rinv[r] = 1.0f / lrow[r];
    #pragma unroll
    for (int dt = 0; dt < 8; ++dt) {
        int d = dt * 16 + m;
        #pragma unroll
        for (int r = 0; r < 8; ++r) {
            int q = q0 + r + 8 * half;
            ob[(size_t)d * NPOS + q] = O[dt][r] * rinv[r];
        }
    }
}

// ---------------------------------------------------------------------------
extern "C" void kernel_launch(void* const* d_in, const int* in_sizes, int n_in,
                              void* d_out, int out_size, void* d_ws, size_t ws_size,
                              hipStream_t stream) {
    const float* x       = (const float*)d_in[0];
    const float* w_theta = (const float*)d_in[1];
    const float* b_theta = (const float*)d_in[2];
    const float* w_phi   = (const float*)d_in[3];
    const float* b_phi   = (const float*)d_in[4];
    const float* w_g     = (const float*)d_in[5];
    const float* b_g     = (const float*)d_in[6];
    float* out = (float*)d_out;

    unsigned short* proj = (unsigned short*)d_ws;          // [3][B][N][128] bf16
    const size_t proj_stride = (size_t)B_SZ * NPOS * C_INT;

    // Kernel 1: 3 projections, 8 batches, 32 position tiles of 128
    dim3 grid1(3 * B_SZ * (NPOS / 128));
    nlb_proj_kernel<<<grid1, 256, 0, stream>>>(x, w_theta, b_theta, w_phi, b_phi,
                                               w_g, b_g, proj);

    // Kernel 2: flash attention over the bf16 projections
    dim3 grid2(B_SZ * (NPOS / 128));
    nlb_attn_kernel<<<grid2, 256, 0, stream>>>(proj,
                                               proj + proj_stride,
                                               proj + 2 * proj_stride,
                                               out);
    (void)in_sizes; (void)n_in; (void)out_size; (void)ws_size;
}
